// MambaLM_87866440941884
// MI455X (gfx1250) — matrix operand order repaired and checked by reading, not compiled, not run on hardware
//
#include <hip/hip_runtime.h>
#include <hip/hip_bf16.h>

// ---------------------------------------------------------------------------
// Mamba LM forward for MI455X (gfx1250, wave32, WMMA).
// Dense GEMMs use v_wmma_f32_16x16x32_bf16; wave tile 32x(JT*16) so each B
// fragment feeds two WMMAs; branch-free inner loops, no EXEC manipulation.
// ---------------------------------------------------------------------------

#define Vv    32000
#define Dm    768
#define NLn   4
#define DIc   1536      // d_inner
#define DIN2  3072      // 2*DI
#define Nst   16        // d_state
#define DCc   4         // d_conv
#define DTR   48        // dt_rank
#define XW    80        // DTR + 2*N
#define Bb    2
#define Ls    1024
#define Mtok  2048      // B*L

typedef __bf16 bf16;
typedef __attribute__((ext_vector_type(16))) __bf16 v16bf;
typedef __attribute__((ext_vector_type(8)))  __bf16 v8bf;
typedef __attribute__((ext_vector_type(8)))  float  v8f;

static __device__ __forceinline__ unsigned short f2bf(float f) {
    unsigned u = __float_as_uint(f);
    u += 0x7FFFu + ((u >> 16) & 1u);     // round-to-nearest-even
    return (unsigned short)(u >> 16);
}
static __device__ __forceinline__ float silu(float x) {
    return x / (1.0f + __expf(-x));
}

// ------------------------------- embedding ---------------------------------
__global__ void k_embed(const int* __restrict__ tok,
                        const float* __restrict__ emb,
                        float* __restrict__ x) {
    int gid = blockIdx.x * blockDim.x + threadIdx.x;
    if (gid >= Mtok * Dm) return;
    int t = gid / Dm, d = gid % Dm;
    x[gid] = emb[(size_t)tok[t] * Dm + d];
}

// ------------------------------ f32 -> bf16 --------------------------------
__global__ void k_cast_bf16(const float* __restrict__ src,
                            unsigned short* __restrict__ dst, int n) {
    int gid = blockIdx.x * blockDim.x + threadIdx.x;
    if (gid < n) dst[gid] = f2bf(src[gid]);
}

// -------------------------------- rmsnorm ----------------------------------
__global__ __launch_bounds__(256) void k_rmsnorm(const float* __restrict__ x,
                                                 const float* __restrict__ w,
                                                 unsigned short* __restrict__ out) {
    __shared__ float red[256];
    int row = blockIdx.x, tid = threadIdx.x;
    const float* xr = x + (size_t)row * Dm;
    float ss = 0.f;
    for (int i = tid; i < Dm; i += 256) { float v = xr[i]; ss += v * v; }
    red[tid] = ss;
    __syncthreads();
    for (int s = 128; s > 0; s >>= 1) {
        if (tid < s) red[tid] += red[tid + s];
        __syncthreads();
    }
    float inv = rsqrtf(red[0] / (float)Dm + 1e-5f);
    unsigned short* orow = out + (size_t)row * Dm;
    for (int i = tid; i < Dm; i += 256) orow[i] = f2bf(xr[i] * inv * w[i]);
}

// ------------------------- WMMA bf16 GEMM ----------------------------------
// C[M,N] = A[M,K] * Bw[N,K]^T (+bias[N]) (+C if beta)
// 256 thr = 8 waves stacked along M; wave tile = 32 x (JT*16):
//   2 M-tiles x JT N-tiles = 2*JT WMMAs per K-step, each B fragment reused
//   twice. Block tile = 256 x (JT*16). Branch-free; EXEC always all-ones.
// Requires: M % 256 == 0, K % 32 == 0, N % (JT*16) == 0.
template <int JT>
__global__ __launch_bounds__(256) void k_gemm(const bf16* __restrict__ A,
                                              const bf16* __restrict__ Bw,
                                              float* __restrict__ C,
                                              int M, int N, int K,
                                              const float* __restrict__ bias,
                                              int beta) {
    const int lane = threadIdx.x & 31;
    const int wave = threadIdx.x >> 5;
    const int half = lane >> 4;
    const int lr   = lane & 15;
    const int m_base = blockIdx.y * 256 + wave * 32;
    const int n_base = blockIdx.x * (JT * 16);

    v8f acc0[JT], acc1[JT];
#pragma unroll
    for (int j = 0; j < JT; ++j) {
        acc0[j] = v8f{0.f, 0.f, 0.f, 0.f, 0.f, 0.f, 0.f, 0.f};
        acc1[j] = v8f{0.f, 0.f, 0.f, 0.f, 0.f, 0.f, 0.f, 0.f};
    }

    // A fragment (16x32 bf16): lane (half,lr) covers K ranges [8h,8h+8) and
    // [16+8h,16+8h+8) of its row -> two contiguous b128 loads per fragment.
    const bf16* arow0 = A + (size_t)(m_base + lr) * K + 8 * half;
    const bf16* arow1 = A + (size_t)(m_base + 16 + lr) * K + 8 * half;
    // B fragment (32x16 bf16): lane (half,lr) covers K ranges [8h,8h+8) and
    // [16+8h,16+8h+8) of column n_base+j*16+lr -> two contiguous b128 loads per fragment.
    const bf16* brow = Bw + (size_t)(n_base + lr) * K + 8 * half;

    for (int k0 = 0; k0 < K; k0 += 32) {
        union { v16bf v; v8bf h[2]; } a0, a1;
        a0.h[0] = *reinterpret_cast<const v8bf*>(arow0 + k0);
        a0.h[1] = *reinterpret_cast<const v8bf*>(arow0 + k0 + 16);
        a1.h[0] = *reinterpret_cast<const v8bf*>(arow1 + k0);
        a1.h[1] = *reinterpret_cast<const v8bf*>(arow1 + k0 + 16);
        if (k0 + 32 < K) {
            __builtin_prefetch(arow0 + k0 + 32, 0, 3);
            __builtin_prefetch(arow1 + k0 + 32, 0, 3);
        }
#pragma unroll
        for (int j = 0; j < JT; ++j) {
            const bf16* bp = brow + (size_t)j * 16 * K + k0;
            union { v16bf v; v8bf h[2]; } bfr;
            bfr.h[0] = *reinterpret_cast<const v8bf*>(bp);
            bfr.h[1] = *reinterpret_cast<const v8bf*>(bp + 16);
            acc0[j] = __builtin_amdgcn_wmma_f32_16x16x32_bf16(
                          false, a0.v, false, bfr.v, (short)0, acc0[j], false, false);
            acc1[j] = __builtin_amdgcn_wmma_f32_16x16x32_bf16(
                          false, a1.v, false, bfr.v, (short)0, acc1[j], false, false);
        }
    }

#pragma unroll
    for (int j = 0; j < JT; ++j) {
        int n = n_base + j * 16 + lr;
        float badd = bias ? bias[n] : 0.f;
#pragma unroll
        for (int r = 0; r < 8; ++r) {
            size_t i0 = (size_t)(m_base + r + 8 * half) * N + n;
            size_t i1 = (size_t)(m_base + 16 + r + 8 * half) * N + n;
            float v0 = acc0[j][r] + badd;
            float v1 = acc1[j][r] + badd;
            if (beta) { v0 += C[i0]; v1 += C[i1]; }
            C[i0] = v0;
            C[i1] = v1;
        }
    }
}

// --------------------- causal depthwise conv + SiLU ------------------------
// xz: (Mtok, 3072); uses first DIc columns. out u (f32) and ub (bf16), (Mtok,DIc)
__global__ void k_conv_silu(const float* __restrict__ xz,
                            const float* __restrict__ cw,
                            const float* __restrict__ cb,
                            float* __restrict__ u,
                            unsigned short* __restrict__ ub) {
    int gid = blockIdx.x * blockDim.x + threadIdx.x;
    if (gid >= Mtok * DIc) return;
    int t = gid / DIc, d = gid % DIc;
    int b = t / Ls, l = t % Ls;
    float acc = cb[d];
#pragma unroll
    for (int j = 0; j < DCc; ++j) {
        int ls = l - (DCc - 1) + j;
        if (ls >= 0)
            acc += cw[d * DCc + j] * xz[(size_t)(b * Ls + ls) * DIN2 + d];
    }
    float s = silu(acc);
    u[gid]  = s;
    ub[gid] = f2bf(s);
}

// ----------------------- dt_proj + softplus --------------------------------
// delta[t,d] = softplus( sum_r xdbl[t,r] * dtw[d,r] + dtb[d] ),  r < 48
__global__ __launch_bounds__(256) void k_dt(const float* __restrict__ xdbl,
                                            const float* __restrict__ dtw,
                                            const float* __restrict__ dtb,
                                            float* __restrict__ delta) {
    __shared__ float s_dt[DTR];
    int t = blockIdx.x, tid = threadIdx.x;
    if (tid < DTR) s_dt[tid] = xdbl[(size_t)t * XW + tid];
    __syncthreads();
    for (int d = tid; d < DIc; d += 256) {
        float acc = dtb[d];
        const float* wr = dtw + (size_t)d * DTR;
#pragma unroll
        for (int r = 0; r < DTR; ++r) acc += s_dt[r] * wr[r];
        float sp = (acc > 20.f) ? acc : __logf(1.f + __expf(acc));
        delta[(size_t)t * DIc + d] = sp;
    }
}

// --------------------------- selective scan --------------------------------
// one thread per (b, d) channel; 16-wide state in registers; sequential in L.
__global__ void k_scan(const float* __restrict__ u,
                       const float* __restrict__ delta,
                       const float* __restrict__ xdbl,
                       const float* __restrict__ A_log,
                       const float* __restrict__ Dv,
                       float* __restrict__ y) {
    int gid = blockIdx.x * blockDim.x + threadIdx.x;
    if (gid >= Bb * DIc) return;
    int b = gid / DIc, d = gid % DIc;
    float A[Nst], h[Nst];
#pragma unroll
    for (int n = 0; n < Nst; ++n) {
        A[n] = -__expf(A_log[d * Nst + n]);
        h[n] = 0.f;
    }
    float Dd = Dv[d];
    for (int l = 0; l < Ls; ++l) {
        int t = b * Ls + l;
        float dt = delta[(size_t)t * DIc + d];
        float uu = u[(size_t)t * DIc + d];
        const float* row = xdbl + (size_t)t * XW;
        float acc = 0.f;
#pragma unroll
        for (int n = 0; n < Nst; ++n) {
            float dA = __expf(dt * A[n]);
            h[n] = dA * h[n] + dt * row[DTR + n] * uu;
            acc += h[n] * row[DTR + Nst + n];
        }
        y[(size_t)t * DIc + d] = acc + uu * Dd;
    }
}

// --------------------------------- gate ------------------------------------
__global__ void k_gate(const float* __restrict__ y,
                       const float* __restrict__ xz,
                       unsigned short* __restrict__ yg) {
    int gid = blockIdx.x * blockDim.x + threadIdx.x;
    if (gid >= Mtok * DIc) return;
    int t = gid / DIc, d = gid % DIc;
    float z = xz[(size_t)t * DIN2 + DIc + d];
    yg[gid] = f2bf(y[gid] * silu(z));
}

// ---------------------------------------------------------------------------
extern "C" void kernel_launch(void* const* d_in, const int* in_sizes, int n_in,
                              void* d_out, int out_size, void* d_ws, size_t ws_size,
                              hipStream_t stream) {
    (void)in_sizes; (void)n_in; (void)out_size; (void)ws_size;

    const int*   tokens    = (const int*)  d_in[0];
    const float* embedding = (const float*)d_in[1];
    const float* W_out_w   = (const float*)d_in[2];
    const float* W_out_b   = (const float*)d_in[3];
    const float* norm_w    = (const float*)d_in[4];
    const float* in_proj_w = (const float*)d_in[5];
    const float* conv_w    = (const float*)d_in[6];
    const float* conv_b    = (const float*)d_in[7];
    const float* x_proj_w  = (const float*)d_in[8];
    const float* dt_proj_w = (const float*)d_in[9];
    const float* dt_proj_b = (const float*)d_in[10];
    const float* A_log     = (const float*)d_in[11];
    const float* Dvec      = (const float*)d_in[12];
    const float* out_proj_w= (const float*)d_in[13];
    float* out = (float*)d_out;

    // ---- workspace bump allocator (256B aligned) ----
    char* wsp = (char*)d_ws;
    auto alloc = [&](size_t bytes) -> void* {
        void* p = (void*)wsp;
        wsp += (bytes + 255) & ~(size_t)255;
        return p;
    };
    unsigned short* bfInW  = (unsigned short*)alloc((size_t)NLn * DIN2 * Dm  * 2);
    unsigned short* bfXpW  = (unsigned short*)alloc((size_t)NLn * XW   * DIc * 2);
    unsigned short* bfOpW  = (unsigned short*)alloc((size_t)NLn * Dm   * DIc * 2);
    unsigned short* bfLW   = (unsigned short*)alloc((size_t)Vv  * Dm          * 2);
    float*          x      = (float*)         alloc((size_t)Mtok * Dm   * 4);
    unsigned short* xnbf   = (unsigned short*)alloc((size_t)Mtok * Dm   * 2);
    float*          xz     = (float*)         alloc((size_t)Mtok * DIN2 * 4);
    float*          ubuf   = (float*)         alloc((size_t)Mtok * DIc  * 4);
    unsigned short* xbbf   = (unsigned short*)alloc((size_t)Mtok * DIc  * 2);
    float*          xdbl   = (float*)         alloc((size_t)Mtok * XW   * 4);
    float*          delta  = (float*)         alloc((size_t)Mtok * DIc  * 4);
    float*          ybuf   = (float*)         alloc((size_t)Mtok * DIc  * 4);
    unsigned short* ygbf   = (unsigned short*)alloc((size_t)Mtok * DIc  * 2);

    const int TB = 256;
    auto blocks = [](long long n, int tb) { return (unsigned)((n + tb - 1) / tb); };

    // ---- one-time weight casts f32 -> bf16 ----
    {
        long long n;
        n = (long long)NLn * DIN2 * Dm;
        k_cast_bf16<<<blocks(n, TB), TB, 0, stream>>>(in_proj_w,  bfInW, (int)n);
        n = (long long)NLn * XW * DIc;
        k_cast_bf16<<<blocks(n, TB), TB, 0, stream>>>(x_proj_w,   bfXpW, (int)n);
        n = (long long)NLn * Dm * DIc;
        k_cast_bf16<<<blocks(n, TB), TB, 0, stream>>>(out_proj_w, bfOpW, (int)n);
        n = (long long)Vv * Dm;
        k_cast_bf16<<<blocks(n, TB), TB, 0, stream>>>(W_out_w,    bfLW,  (int)n);
    }

    // ---- embedding gather ----
    k_embed<<<blocks((long long)Mtok * Dm, TB), TB, 0, stream>>>(tokens, embedding, x);

    // ---- layers ----
    for (int li = 0; li < NLn; ++li) {
        const unsigned short* wIn = bfInW + (size_t)li * DIN2 * Dm;
        const unsigned short* wXp = bfXpW + (size_t)li * XW * DIc;
        const unsigned short* wOp = bfOpW + (size_t)li * Dm * DIc;

        k_rmsnorm<<<Mtok, 256, 0, stream>>>(x, norm_w + (size_t)li * Dm, xnbf);

        // in_proj: (2048 x 768) * (3072 x 768)^T   -> JT=4, grid 48x8
        k_gemm<4><<<dim3(DIN2 / 64, Mtok / 256), 256, 0, stream>>>(
            (const bf16*)xnbf, (const bf16*)wIn, xz, Mtok, DIN2, Dm, nullptr, 0);

        k_conv_silu<<<blocks((long long)Mtok * DIc, TB), TB, 0, stream>>>(
            xz, conv_w + (size_t)li * DIc * DCc, conv_b + (size_t)li * DIc, ubuf, xbbf);

        // x_proj: (2048 x 1536) * (80 x 1536)^T    -> JT=5 (80 = 5*16), grid 1x8
        k_gemm<5><<<dim3(XW / 80, Mtok / 256), 256, 0, stream>>>(
            (const bf16*)xbbf, (const bf16*)wXp, xdbl, Mtok, XW, DIc, nullptr, 0);

        k_dt<<<Mtok, 256, 0, stream>>>(
            xdbl, dt_proj_w + (size_t)li * DIc * DTR, dt_proj_b + (size_t)li * DIc, delta);

        k_scan<<<blocks((long long)Bb * DIc, TB), TB, 0, stream>>>(
            ubuf, delta, xdbl, A_log + (size_t)li * DIc * Nst, Dvec + (size_t)li * DIc, ybuf);

        k_gate<<<blocks((long long)Mtok * DIc, TB), TB, 0, stream>>>(ybuf, xz, ygbf);

        // out_proj + residual: x += (2048 x 1536) * (768 x 1536)^T  -> JT=4
        k_gemm<4><<<dim3(Dm / 64, Mtok / 256), 256, 0, stream>>>(
            (const bf16*)ygbf, (const bf16*)wOp, x, Mtok, Dm, DIc, nullptr, 1);
    }

    // ---- logits: (2048 x 768) * (32000 x 768)^T + bias  -> JT=4, grid 500x8
    k_cast_bf16<<<blocks((long long)Mtok * Dm, TB), TB, 0, stream>>>(
        x, xnbf, Mtok * Dm);
    k_gemm<4><<<dim3(Vv / 64, Mtok / 256), 256, 0, stream>>>(
        (const bf16*)xnbf, (const bf16*)bfLW, out, Mtok, Vv, Dm, W_out_b, 0);
}
